// HGTLayer_11269994185388
// MI455X (gfx1250) — compile-verified, matrix operand order checked by reading, not run on hardware
//
#include <hip/hip_runtime.h>
#include <hip/hip_bf16.h>
#include <math.h>

#define N_NODES 50000
#define IN_DIM  256
#define OUT_DIM 256
#define R_REL   4
#define E_EDGE  50000
#define H_HEADS 4
#define DK_DIM  64

typedef __attribute__((ext_vector_type(16))) __bf16 v16bf;
typedef __attribute__((ext_vector_type(8)))  __bf16 v8bf;
typedef __attribute__((ext_vector_type(8)))  float  v8f;

__device__ __forceinline__ unsigned short f2bf_u(float f) {
  unsigned int u = __float_as_uint(f);
  return (unsigned short)((u + 0x7FFFu + ((u >> 16) & 1u)) >> 16);   // RNE
}
__device__ __forceinline__ float bf2f(unsigned short s) {
  return __uint_as_float(((unsigned int)s) << 16);
}
__device__ __forceinline__ void atomicMaxF(float* addr, float v) {
  if (v >= 0.0f) atomicMax((int*)addr, __float_as_int(v));
  else           atomicMin((unsigned int*)addr, __float_as_uint(v));
}

// ---------------------------------------------------------------------------
// Generic bf16 WMMA GEMM:  Y[M x Ncols] = A[M x K] * B^T + bias, optional
// sigmoid-skip residual blend.  B given row-major as [Ncols x K] (weight
// layout W[o][i]), so each lane streams a contiguous weight row.
// 128 threads = 4 waves, one 16x16 output tile per wave.
// ---------------------------------------------------------------------------
__global__ void wmma_gemm(const unsigned short* __restrict__ A, int lda,
                          const unsigned short* __restrict__ B, int ldb,
                          const float* __restrict__ bias,
                          float* __restrict__ outF,
                          unsigned short* __restrict__ outB, int ldo,
                          int Kdim, int ntilesN,
                          const float* __restrict__ skipPtr,
                          const float* __restrict__ resid, int ldr)
{
  const int wid   = threadIdx.x >> 5;
  const int lane  = threadIdx.x & 31;
  const int gtile = blockIdx.x * 4 + wid;
  const int tn = gtile % ntilesN;
  const int tm = gtile / ntilesN;
  const int m0 = tm * 16, n0 = tn * 16;
  const int row  = lane & 15;
  const int half = lane >> 4;

  // A lane: row m0+row, K slots {half*8..+7, 16+half*8..+7} per 32-wide k-step
  const unsigned short* ap = A + (size_t)(m0 + row) * lda + half * 8;
  const unsigned short* bp = B + (size_t)(n0 + row) * ldb + half * 8;

  v8f c = {};
  for (int k0 = 0; k0 < Kdim; k0 += 32) {
    v8bf alo = *(const v8bf*)(ap + k0);
    v8bf ahi = *(const v8bf*)(ap + k0 + 16);
    v8bf blo = *(const v8bf*)(bp + k0);
    v8bf bhi = *(const v8bf*)(bp + k0 + 16);
    v16bf a, b;
#pragma unroll
    for (int i = 0; i < 8; ++i) {
      a[i] = alo[i]; a[i + 8] = ahi[i];
      b[i] = blo[i]; b[i + 8] = bhi[i];
    }
    c = __builtin_amdgcn_wmma_f32_16x16x32_bf16(
            false, a, false, b, (short)0, c, false, false);
  }

  float alpha = 1.0f, beta = 0.0f;
  if (skipPtr) {
    float s = *skipPtr;
    alpha = 1.0f / (1.0f + expf(-s));
    beta  = 1.0f - alpha;
  }
  const int col = n0 + row;
  const float bv = bias ? bias[col] : 0.0f;
#pragma unroll
  for (int j = 0; j < 8; ++j) {
    const int rrow = m0 + j + half * 8;
    float v = c[j] + bv;
    if (resid) v = v * alpha + beta * resid[(size_t)rrow * ldr + col];
    if (outF) outF[(size_t)rrow * ldo + col] = v;
    if (outB) outB[(size_t)rrow * ldo + col] = f2bf_u(v);
  }
}

// ---------------------------------------------------------------------------
// Elementwise helpers
// ---------------------------------------------------------------------------
__global__ void cvt_bf16(const float* __restrict__ in,
                         unsigned short* __restrict__ out, unsigned int n) {
  unsigned int i = blockIdx.x * blockDim.x + threadIdx.x;
  if (i < n) out[i] = f2bf_u(in[i]);
}

// rel [R,H,DK,DK] (d,f) -> transposed bf16 [R*H][f][d] so GEMM's "weight row"
// addressing (row = output column f, contiguous over d) applies directly.
__global__ void cvt_rel_t(const float* __restrict__ in,
                          unsigned short* __restrict__ out) {
  unsigned int idx = blockIdx.x * blockDim.x + threadIdx.x;
  if (idx >= R_REL * H_HEADS * DK_DIM * DK_DIM) return;
  unsigned int rh  = idx / (DK_DIM * DK_DIM);
  unsigned int rem = idx - rh * (DK_DIM * DK_DIM);
  unsigned int d = rem >> 6, f = rem & 63;
  out[rh * (DK_DIM * DK_DIM) + f * DK_DIM + d] = f2bf_u(in[idx]);
}

__global__ void fill_f32(float* __restrict__ p, float v, unsigned int n) {
  unsigned int i = blockIdx.x * blockDim.x + threadIdx.x;
  if (i < n) p[i] = v;
}

__global__ void t_scale(const float* __restrict__ t,
                        unsigned short* __restrict__ tb, unsigned int n) {
  unsigned int i = blockIdx.x * blockDim.x + threadIdx.x;
  if (i < n) tb[i] = f2bf_u(t[i] * 0.25f);   // mean over R=4 relations
}

// ---------------------------------------------------------------------------
// Edge passes.  One wave32 per (r,e,h): lane l owns channels d=l and d=l+32,
// so every vmem burst (loads and atomics) is a fully coalesced 128B segment
// across the wave instead of a 64-long serial chain per thread.
// R*E*H = 800000 waves = 100000 blocks of 8 waves.
// ---------------------------------------------------------------------------
__global__ void edge_att_kernel(const int* __restrict__ esrc,
                                const int* __restrict__ edst,
                                const unsigned short* __restrict__ KAb,
                                const float* __restrict__ Qf,
                                const float* __restrict__ pri,
                                float* __restrict__ att,
                                float* __restrict__ mmax) {
  const unsigned int weh  = blockIdx.x * 8 + (threadIdx.x >> 5);
  const int          lane = threadIdx.x & 31;
  if (weh >= (unsigned)R_REL * E_EDGE * H_HEADS) return;
  const int h = weh & (H_HEADS - 1);
  const unsigned int eh = weh >> 2;
  const int r = eh / E_EDGE;
  const int e = eh - (unsigned)r * E_EDGE;
  const int s = esrc[r * E_EDGE + e];
  const int d = edst[r * E_EDGE + e];
  const unsigned short* ka = KAb + ((size_t)r * N_NODES + s) * OUT_DIM + h * DK_DIM;
  const float*          q  = Qf  + (size_t)d * OUT_DIM + h * DK_DIM;

  float acc = bf2f(ka[lane])      * q[lane]
            + bf2f(ka[lane + 32]) * q[lane + 32];
#pragma unroll
  for (int m = 16; m >= 1; m >>= 1) acc += __shfl_xor(acc, m, 32);

  if (lane == 0) {
    float a = acc * pri[r * H_HEADS + h] * 0.125f;   // / sqrt(DK=64)
    att[weh] = a;
    atomicMaxF(&mmax[((size_t)r * N_NODES + d) * H_HEADS + h], a);
  }
}

__global__ void edge_exp_kernel(const int* __restrict__ edst,
                                float* __restrict__ att,
                                const float* __restrict__ mmax,
                                float* __restrict__ den) {
  unsigned int idx = blockIdx.x * blockDim.x + threadIdx.x;
  if (idx >= (unsigned)R_REL * E_EDGE * H_HEADS) return;
  int h = idx & (H_HEADS - 1);
  unsigned int eh = idx >> 2;
  int r = eh / E_EDGE;
  int e = eh - (unsigned)r * E_EDGE;
  int d = edst[r * E_EDGE + e];
  float m = mmax[((size_t)r * N_NODES + d) * H_HEADS + h];
  float ex = expf(att[idx] - m);
  att[idx] = ex;
  atomicAdd(&den[((size_t)r * N_NODES + d) * H_HEADS + h], ex);
}

__global__ void edge_scatter_kernel(const int* __restrict__ esrc,
                                    const int* __restrict__ edst,
                                    const float* __restrict__ att,
                                    const float* __restrict__ den,
                                    const unsigned short* __restrict__ VMb,
                                    float* __restrict__ tacc) {
  const unsigned int weh  = blockIdx.x * 8 + (threadIdx.x >> 5);
  const int          lane = threadIdx.x & 31;
  if (weh >= (unsigned)R_REL * E_EDGE * H_HEADS) return;
  const int h = weh & (H_HEADS - 1);
  const unsigned int eh = weh >> 2;
  const int r = eh / E_EDGE;
  const int e = eh - (unsigned)r * E_EDGE;
  const int s = esrc[r * E_EDGE + e];
  const int d = edst[r * E_EDGE + e];
  const float w = att[weh] /
      fmaxf(den[((size_t)r * N_NODES + d) * H_HEADS + h], 1e-9f);
  const unsigned short* vm = VMb + ((size_t)r * N_NODES + s) * OUT_DIM + h * DK_DIM;
  float* t = tacc + (size_t)d * OUT_DIM + h * DK_DIM;
  atomicAdd(&t[lane],      w * bf2f(vm[lane]));
  atomicAdd(&t[lane + 32], w * bf2f(vm[lane + 32]));
}

// ---------------------------------------------------------------------------
extern "C" void kernel_launch(void* const* d_in, const int* in_sizes, int n_in,
                              void* d_out, int out_size, void* d_ws, size_t ws_size,
                              hipStream_t stream) {
  (void)in_sizes; (void)n_in; (void)out_size; (void)ws_size;
  const float* x    = (const float*)d_in[0];
  const int*   esrc = (const int*)d_in[1];
  const int*   edst = (const int*)d_in[2];
  const float* Wk   = (const float*)d_in[3];
  const float* bk   = (const float*)d_in[4];
  const float* Wq   = (const float*)d_in[5];
  const float* bq   = (const float*)d_in[6];
  const float* Wv   = (const float*)d_in[7];
  const float* bvp  = (const float*)d_in[8];
  const float* Wa   = (const float*)d_in[9];
  const float* ba   = (const float*)d_in[10];
  const float* pri  = (const float*)d_in[11];
  const float* ratt = (const float*)d_in[12];
  const float* rmsg = (const float*)d_in[13];
  const float* skip = (const float*)d_in[14];
  float* out = (float*)d_out;

  char* ws = (char*)d_ws;
  size_t off = 0;
  auto take = [&](size_t bytes) -> char* {
    char* p = ws + off;
    off = (off + bytes + 255) & ~(size_t)255;
    return p;
  };
  const size_t NF = (size_t)N_NODES * OUT_DIM;            // 12.8M elems
  unsigned short* xb  = (unsigned short*)take((size_t)N_NODES * IN_DIM * 2);
  unsigned short* Wkb = (unsigned short*)take(65536 * 2);
  unsigned short* Wqb = (unsigned short*)take(65536 * 2);
  unsigned short* Wvb = (unsigned short*)take(65536 * 2);
  unsigned short* Wab = (unsigned short*)take(65536 * 2);
  unsigned short* Atb = (unsigned short*)take(65536 * 2); // rel_att^T
  unsigned short* Mtb = (unsigned short*)take(65536 * 2); // rel_msg^T
  unsigned short* Kb  = (unsigned short*)take(NF * 2);
  unsigned short* Vb  = (unsigned short*)take(NF * 2);
  float*          Qf  = (float*)take(NF * 4);
  unsigned short* KAb = (unsigned short*)take((size_t)R_REL * NF * 2);
  unsigned short* VMb = (unsigned short*)take((size_t)R_REL * NF * 2);
  float* attb = (float*)take((size_t)R_REL * E_EDGE * H_HEADS * 4);
  float* mmax = (float*)take((size_t)R_REL * N_NODES * H_HEADS * 4);
  float* den  = (float*)take((size_t)R_REL * N_NODES * H_HEADS * 4);
  float* tacc = (float*)take(NF * 4);
  unsigned short* tb = (unsigned short*)take(NF * 2);

  const unsigned int nXb   = N_NODES * IN_DIM;
  const unsigned int nW    = 65536;
  const unsigned int nRNH  = R_REL * N_NODES * H_HEADS;   // 800000
  const unsigned int nREH  = R_REL * E_EDGE * H_HEADS;    // 800000

  // 1) precision downcast / transposes
  cvt_bf16<<<(nXb + 255) / 256, 256, 0, stream>>>(x,  xb,  nXb);
  cvt_bf16<<<(nW + 255) / 256, 256, 0, stream>>>(Wk, Wkb, nW);
  cvt_bf16<<<(nW + 255) / 256, 256, 0, stream>>>(Wq, Wqb, nW);
  cvt_bf16<<<(nW + 255) / 256, 256, 0, stream>>>(Wv, Wvb, nW);
  cvt_bf16<<<(nW + 255) / 256, 256, 0, stream>>>(Wa, Wab, nW);
  cvt_rel_t<<<(nW + 255) / 256, 256, 0, stream>>>(ratt, Atb);
  cvt_rel_t<<<(nW + 255) / 256, 256, 0, stream>>>(rmsg, Mtb);

  // 2) init accumulators (ws is poisoned; must re-init every call)
  fill_f32<<<(nRNH + 255) / 256, 256, 0, stream>>>(mmax, -3.0e38f, nRNH);
  fill_f32<<<(nRNH + 255) / 256, 256, 0, stream>>>(den,   0.0f,    nRNH);
  fill_f32<<<((unsigned)NF + 255) / 256, 256, 0, stream>>>(tacc, 0.0f, (unsigned)NF);

  // 3) K/Q/V projections: [50000x256] @ W^T + b.  3125*16 tiles / 4 waves.
  const int gemmBlocks = (N_NODES / 16) * (OUT_DIM / 16) / 4;   // 12500
  wmma_gemm<<<gemmBlocks, 128, 0, stream>>>(xb, IN_DIM, Wkb, IN_DIM, bk,
      nullptr, Kb, OUT_DIM, IN_DIM, OUT_DIM / 16, nullptr, nullptr, 0);
  wmma_gemm<<<gemmBlocks, 128, 0, stream>>>(xb, IN_DIM, Wqb, IN_DIM, bq,
      Qf, nullptr, OUT_DIM, IN_DIM, OUT_DIM / 16, nullptr, nullptr, 0);
  wmma_gemm<<<gemmBlocks, 128, 0, stream>>>(xb, IN_DIM, Wvb, IN_DIM, bvp,
      nullptr, Vb, OUT_DIM, IN_DIM, OUT_DIM / 16, nullptr, nullptr, 0);

  // 4) per-(r,h) relation transforms as dense GEMMs (E==N, same FLOPs):
  //    KA[r,:,h,:] = K[:,h,:] @ rel_att[r,h];  VM likewise with rel_msg
  const int relBlocks = (N_NODES / 16) * (DK_DIM / 16) / 4;     // 3125
  for (int r = 0; r < R_REL; ++r) {
    for (int h = 0; h < H_HEADS; ++h) {
      const unsigned short* At = Atb + (size_t)(r * H_HEADS + h) * DK_DIM * DK_DIM;
      const unsigned short* Mt = Mtb + (size_t)(r * H_HEADS + h) * DK_DIM * DK_DIM;
      wmma_gemm<<<relBlocks, 128, 0, stream>>>(
          Kb + h * DK_DIM, OUT_DIM, At, DK_DIM, nullptr,
          nullptr, KAb + (size_t)r * NF + h * DK_DIM, OUT_DIM,
          DK_DIM, DK_DIM / 16, nullptr, nullptr, 0);
      wmma_gemm<<<relBlocks, 128, 0, stream>>>(
          Vb + h * DK_DIM, OUT_DIM, Mt, DK_DIM, nullptr,
          nullptr, VMb + (size_t)r * NF + h * DK_DIM, OUT_DIM,
          DK_DIM, DK_DIM / 16, nullptr, nullptr, 0);
    }
  }

  // 5) segment softmax over incoming edges + weighted scatter (3 passes)
  const int waveBlocks = nREH / 8;                              // 100000
  const int edgeBlocks = (nREH + 255) / 256;                    // 3125
  edge_att_kernel<<<waveBlocks, 256, 0, stream>>>(esrc, edst, KAb, Qf, pri, attb, mmax);
  edge_exp_kernel<<<edgeBlocks, 256, 0, stream>>>(edst, attb, mmax, den);
  edge_scatter_kernel<<<waveBlocks, 256, 0, stream>>>(esrc, edst, attb, den, VMb, tacc);

  // 6) mean over relations, downcast, final GEMM with sigmoid-skip blend
  t_scale<<<((unsigned)NF + 255) / 256, 256, 0, stream>>>(tacc, tb, (unsigned)NF);
  wmma_gemm<<<gemmBlocks, 128, 0, stream>>>(tb, OUT_DIM, Wab, OUT_DIM, ba,
      out, nullptr, OUT_DIM, OUT_DIM, OUT_DIM / 16, skip, x, IN_DIM);
}
